// GATDecoder_31190052503740
// MI455X (gfx1250) — matrix operand order repaired and checked by reading, not compiled, not run on hardware
//
#include <hip/hip_runtime.h>

// ---------------- problem constants (from reference) ----------------
#define N_NODES   100000
#define N_EDGES   3200000
#define ETOT      (N_EDGES + N_NODES)   // with self loops
#define IN_CH     256
#define OUT_CH    64
#define HEADS     2
#define OUT2      (HEADS * OUT_CH)      // 128
#define NEG_SLOPE 0.2f

typedef __attribute__((ext_vector_type(16))) _Float16 v16h;
typedef __attribute__((ext_vector_type(8)))  _Float16 v8h;
typedef __attribute__((ext_vector_type(8)))  float    v8f;

// ---------------- helpers ----------------
__device__ __forceinline__ float lrelu(float v) {
    return v >= 0.0f ? v : NEG_SLOPE * v;
}
// monotonic float<->uint key for atomicMax on floats (handles negatives)
__device__ __forceinline__ unsigned fkey(float f) {
    unsigned u = __float_as_uint(f);
    return (u & 0x80000000u) ? ~u : (u | 0x80000000u);
}
__device__ __forceinline__ float funkey(unsigned k) {
    return __uint_as_float((k & 0x80000000u) ? (k & 0x7fffffffu) : ~k);
}
__device__ __forceinline__ void edge_sd(const long long* __restrict__ ei, int e,
                                        int& s, int& d) {
    if (e < N_EDGES) { s = (int)ei[e]; d = (int)ei[(size_t)N_EDGES + e]; }
    else             { s = d = e - N_EDGES; }      // self loop
}

// ---------------- kernel 1: pack W (fp32 256x128) into B-fragment order f16 ----
// Whp[(((nt*8+kt)*2+j)*32 + lane)*8 + i] = W[(kt*32 + j*16 + (lane>>4)*8 + i)*128 + nt*16 + (lane&15)]
__global__ void pack_w_kernel(const float* __restrict__ W, _Float16* __restrict__ Whp) {
    int t = blockIdx.x * blockDim.x + threadIdx.x;
    if (t >= IN_CH * OUT2) return;
    int i    = t & 7;
    int lane = (t >> 3) & 31;
    int j    = (t >> 8) & 1;
    int kt   = (t >> 9) & 7;
    int nt   = (t >> 12) & 7;
    int k = kt * 32 + j * 16 + (lane >> 4) * 8 + i;
    int n = nt * 16 + (lane & 15);
    Whp[t] = (_Float16)W[(size_t)k * OUT2 + n];
}

// ---------------- kernel 2: init out=bias, zero softmax stats ----------------
__global__ void init_kernel(const float* __restrict__ bias, float* __restrict__ out,
                            unsigned* __restrict__ seg_max_u, float* __restrict__ seg_sum) {
    int t = blockIdx.x * blockDim.x + threadIdx.x;
    if (t < N_NODES * OUT_CH) out[t] = bias[t & (OUT_CH - 1)];
    if (t < N_NODES * HEADS) { seg_max_u[t] = 0u; seg_sum[t] = 0.0f; }
}

// ---------------- kernel 3: WMMA GEMM  x = z @ W  (f16 in, f32 out) ----------
// block = 256 thr = 8 waves; block computes rows [m0,m0+16) x all 128 cols;
// wave w computes the 16x16 tile at cols [16w,16w+16).
#define LDSS (IN_CH + 8)   // f16 row stride in LDS: 528 B, keeps 16-B alignment
__global__ __launch_bounds__(256) void gemm_kernel(const float* __restrict__ z,
                                                   const _Float16* __restrict__ Whp,
                                                   float* __restrict__ x) {
    __shared__ __align__(16) _Float16 As[16 * LDSS];
    const int m0   = blockIdx.x * 16;
    const int tid  = threadIdx.x;
    const int wave = tid >> 5;          // n-tile 0..7
    const int lane = tid & 31;

    // cooperative load of A tile: 16 rows x 256 f32 -> f16 LDS (16 f32 / thread)
    {
        int r = tid >> 4;
        int c = (tid & 15) * 16;
        const float4* zp = (const float4*)(z + (size_t)(m0 + r) * IN_CH + c);
        _Float16* dp = As + r * LDSS + c;
#pragma unroll
        for (int jj = 0; jj < 4; ++jj) {
            float4 v = zp[jj];
            dp[jj * 4 + 0] = (_Float16)v.x;
            dp[jj * 4 + 1] = (_Float16)v.y;
            dp[jj * 4 + 2] = (_Float16)v.z;
            dp[jj * 4 + 3] = (_Float16)v.w;
        }
    }
    __syncthreads();

    const int mrow  = lane & 15;
    const int khalf = lane >> 4;
    v8f acc = {};
#pragma unroll
    for (int kt = 0; kt < 8; ++kt) {
        // A fragment: two contiguous 16-B chunks per lane (ISA 16-bit A 16x32 layout)
        const _Float16* ap = As + mrow * LDSS + kt * 32 + khalf * 8;
        v8h alo = *(const v8h*)ap;
        v8h ahi = *(const v8h*)(ap + 16);
        v16h a = __builtin_shufflevector(alo, ahi,
                 0,1,2,3,4,5,6,7,8,9,10,11,12,13,14,15);
        // B fragment: pre-packed, two coalesced 16-B loads per lane
        const v8h* wp = (const v8h*)Whp;
        v8h blo = wp[((wave * 8 + kt) * 2 + 0) * 32 + lane];
        v8h bhi = wp[((wave * 8 + kt) * 2 + 1) * 32 + lane];
        v16h b = __builtin_shufflevector(blo, bhi,
                 0,1,2,3,4,5,6,7,8,9,10,11,12,13,14,15);
        acc = __builtin_amdgcn_wmma_f32_16x16x32_f16(
                  false, a, false, b, (short)0, acc, false, false);
    }

    // C/D layout: VGPR r -> row m0 + (lane>>4)*8 + r, col 16*wave + (lane&15)
    const int n = wave * 16 + (lane & 15);
    const int mb = m0 + (lane >> 4) * 8;
#pragma unroll
    for (int r = 0; r < 8; ++r)
        x[(size_t)(mb + r) * OUT2 + n] = acc[r];
}

// ---------------- kernel 4: per-node attention scalars -----------------------
__global__ void att_kernel(const float* __restrict__ x,
                           const float* __restrict__ att_s, const float* __restrict__ att_d,
                           float* __restrict__ a_src, float* __restrict__ a_dst) {
    int t = blockIdx.x * blockDim.x + threadIdx.x;
    if (t >= N_NODES * HEADS) return;
    int n = t >> 1, h = t & 1;
    const float* xr = x + (size_t)n * OUT2 + h * OUT_CH;
    const float* as = att_s + h * OUT_CH;
    const float* ad = att_d + h * OUT_CH;
    float s = 0.0f, d = 0.0f;
#pragma unroll 4
    for (int c = 0; c < OUT_CH; c += 4) {
        float4 xv = *(const float4*)(xr + c);
        float4 sv = *(const float4*)(as + c);
        float4 dv = *(const float4*)(ad + c);
        s += xv.x * sv.x + xv.y * sv.y + xv.z * sv.z + xv.w * sv.w;
        d += xv.x * dv.x + xv.y * dv.y + xv.z * dv.z + xv.w * dv.w;
    }
    a_src[t] = s; a_dst[t] = d;
}

// ---------------- kernel 5: segment max over edges (pass 1) ------------------
__global__ void edge_max_kernel(const long long* __restrict__ ei,
                                const float* __restrict__ a_src, const float* __restrict__ a_dst,
                                unsigned* __restrict__ seg_max_u) {
    int e = blockIdx.x * blockDim.x + threadIdx.x;
    if (e >= ETOT) return;
    int s, d; edge_sd(ei, e, s, d);
    float l0 = lrelu(a_src[2 * s + 0] + a_dst[2 * d + 0]);
    float l1 = lrelu(a_src[2 * s + 1] + a_dst[2 * d + 1]);
    atomicMax(&seg_max_u[2 * d + 0], fkey(l0));
    atomicMax(&seg_max_u[2 * d + 1], fkey(l1));
}

// ---------------- kernel 6: segment exp-sum (pass 2) -------------------------
__global__ void edge_sum_kernel(const long long* __restrict__ ei,
                                const float* __restrict__ a_src, const float* __restrict__ a_dst,
                                const unsigned* __restrict__ seg_max_u,
                                float* __restrict__ seg_sum) {
    int e = blockIdx.x * blockDim.x + threadIdx.x;
    if (e >= ETOT) return;
    int s, d; edge_sd(ei, e, s, d);
    float l0 = lrelu(a_src[2 * s + 0] + a_dst[2 * d + 0]);
    float l1 = lrelu(a_src[2 * s + 1] + a_dst[2 * d + 1]);
    float e0 = __expf(l0 - funkey(seg_max_u[2 * d + 0]));
    float e1 = __expf(l1 - funkey(seg_max_u[2 * d + 1]));
    atomicAdd(&seg_sum[2 * d + 0], e0);
    atomicAdd(&seg_sum[2 * d + 1], e1);
}

// ---------------- kernel 7: alpha-weighted scatter (head-mean fused) ---------
// one wave per edge; lane L handles output channels 2L, 2L+1
__global__ __launch_bounds__(256) void scatter_kernel(const long long* __restrict__ ei,
                                                      const float* __restrict__ x,
                                                      const float* __restrict__ a_src,
                                                      const float* __restrict__ a_dst,
                                                      const unsigned* __restrict__ seg_max_u,
                                                      const float* __restrict__ seg_sum,
                                                      float* __restrict__ out) {
    int e = blockIdx.x * 8 + (threadIdx.x >> 5);
    if (e >= ETOT) return;
    const int lane = threadIdx.x & 31;
    int s, d; edge_sd(ei, e, s, d);
    float l0 = lrelu(a_src[2 * s + 0] + a_dst[2 * d + 0]);
    float l1 = lrelu(a_src[2 * s + 1] + a_dst[2 * d + 1]);
    float alpha0 = __expf(l0 - funkey(seg_max_u[2 * d + 0])) / seg_sum[2 * d + 0];
    float alpha1 = __expf(l1 - funkey(seg_max_u[2 * d + 1])) / seg_sum[2 * d + 1];
    float2 x0 = ((const float2*)(x + (size_t)s * OUT2))[lane];            // head 0
    float2 x1 = ((const float2*)(x + (size_t)s * OUT2 + OUT_CH))[lane];   // head 1
    float r0 = 0.5f * (alpha0 * x0.x + alpha1 * x1.x);
    float r1 = 0.5f * (alpha0 * x0.y + alpha1 * x1.y);
    float* op = out + (size_t)d * OUT_CH + lane * 2;
    atomicAdd(op + 0, r0);
    atomicAdd(op + 1, r1);
}

// ---------------- launch ----------------
extern "C" void kernel_launch(void* const* d_in, const int* in_sizes, int n_in,
                              void* d_out, int out_size, void* d_ws, size_t ws_size,
                              hipStream_t stream) {
    const float*     z     = (const float*)d_in[0];
    const long long* ei    = (const long long*)d_in[1];   // int64 (2, E)
    const float*     W     = (const float*)d_in[2];
    const float*     att_s = (const float*)d_in[3];
    const float*     att_d = (const float*)d_in[4];
    const float*     bias  = (const float*)d_in[5];
    float*           out   = (float*)d_out;

    // workspace carve-up (all sizes are multiples of 256 B)
    char* w = (char*)d_ws;
    float*     x         = (float*)w;      w += (size_t)N_NODES * OUT2 * sizeof(float);   // 51.2 MB
    _Float16*  Whp       = (_Float16*)w;   w += (size_t)IN_CH * OUT2 * sizeof(_Float16);  // 64 KB
    float*     a_src     = (float*)w;      w += (size_t)N_NODES * HEADS * sizeof(float);
    float*     a_dst     = (float*)w;      w += (size_t)N_NODES * HEADS * sizeof(float);
    unsigned*  seg_max_u = (unsigned*)w;   w += (size_t)N_NODES * HEADS * sizeof(unsigned);
    float*     seg_sum   = (float*)w;      w += (size_t)N_NODES * HEADS * sizeof(float);

    pack_w_kernel<<<(IN_CH * OUT2 + 255) / 256, 256, 0, stream>>>(W, Whp);
    init_kernel<<<(N_NODES * OUT_CH + 255) / 256, 256, 0, stream>>>(bias, out, seg_max_u, seg_sum);
    gemm_kernel<<<N_NODES / 16, 256, 0, stream>>>(z, Whp, x);
    att_kernel<<<(N_NODES * HEADS + 255) / 256, 256, 0, stream>>>(x, att_s, att_d, a_src, a_dst);
    edge_max_kernel<<<(ETOT + 255) / 256, 256, 0, stream>>>(ei, a_src, a_dst, seg_max_u);
    edge_sum_kernel<<<(ETOT + 255) / 256, 256, 0, stream>>>(ei, a_src, a_dst, seg_max_u, seg_sum);
    scatter_kernel<<<(ETOT + 7) / 8, 256, 0, stream>>>(ei, x, a_src, a_dst, seg_max_u, seg_sum, out);
}
